// Net3D_58188216926997
// MI455X (gfx1250) — compile-verified, matrix operand order checked by reading, not run on hardware
//
#include <hip/hip_runtime.h>

// ---------------------------------------------------------------------------
// Net3D forward for MI455X (gfx1250, wave32, WMMA bf16 16x16x32 -> f32 acc)
// A-tiles staged into LDS with gfx1250 async global->LDS copies (ASYNCcnt),
// double-buffered so the copy of tile k+1 overlaps the WMMAs of tile k.
// ---------------------------------------------------------------------------
typedef long long i64;
typedef __attribute__((ext_vector_type(4)))  float   f32x4;
typedef __attribute__((ext_vector_type(8)))  float   v8f;
typedef __attribute__((ext_vector_type(8)))  __bf16  bf16x8;
typedef __attribute__((ext_vector_type(16))) __bf16  v16bf;

#define HID 128
#define EPS 1e-5f

__device__ __forceinline__ int kmap_lane(int lane, int j) {
    // 16-bit A/B fragment K mapping (ISA 7.12.2): lanes 0-15 hold K {0..7,16..23},
    // lanes 16-31 hold K {8..15,24..31}.
    int kh = (lane >> 4) * 8;
    return (j < 8) ? (kh + j) : (16 + kh + (j - 8));
}

// gfx1250 async global->LDS 16-byte copy (tracked by ASYNCcnt).
__device__ __forceinline__ void async_ld_b128(unsigned ldsOff, const void* gptr) {
    asm volatile("global_load_async_to_lds_b128 %0, %1, off"
                 :: "v"(ldsOff), "v"((unsigned long long)gptr)
                 : "memory");
}
__device__ __forceinline__ void wait_async0() {
    asm volatile("s_wait_asynccnt 0x0" ::: "memory");
}

// --- Convert f32 weight [K x 128] row-major into WMMA-B fragment layout -----
__global__ void wconv_frag(const float* __restrict__ W, __bf16* __restrict__ out, int K) {
    int tid = blockIdx.x * blockDim.x + threadIdx.x;
    int total = (K >> 5) * 8 * 32;              // (ksteps)*(nfrags)*(lanes)
    if (tid >= total) return;
    int lane  = tid & 31;
    int fid   = tid >> 5;
    int nfrag = fid & 7;
    int ks    = fid >> 3;
    int col   = nfrag * 16 + (lane & 15);
    __bf16* o = out + (i64)tid * 16;
    #pragma unroll
    for (int j = 0; j < 16; ++j) {
        int k = ks * 32 + kmap_lane(lane, j);
        o[j] = (__bf16)W[(i64)k * HID + col];
    }
}

// --- Generic bf16 GEMM: C[rows,128] = act(A[rows,K] @ Bfrag + bias) ---------
// EPI: 0 = store f32   1 = relu -> bf16   2 = SiLU -> f32   3 = resid + -> f32
template <int EPI>
__global__ __launch_bounds__(256) void gemm_bf16(
    const __bf16* __restrict__ A, const __bf16* __restrict__ Bf,
    const float* __restrict__ bias, const float* __restrict__ resid,
    float* __restrict__ outF, __bf16* __restrict__ outB, int rows, int K) {
    __shared__ alignas(16) __bf16 As[2][64 * 40];
    const int tid  = threadIdx.x;
    const int wave = tid >> 5, lane = tid & 31;
    const int m0 = (wave >> 1) * 16;   // 0,16,32,48
    const int n0 = (wave & 1) * 64;    // 0,64
    const int r  = tid >> 2;           // 0..63 (fill row)
    const int c8 = (tid & 3) * 8;      // fill col offset
    const i64 grow  = (i64)blockIdx.x * 64 + r;
    const i64 growc = grow < rows ? grow : (i64)rows - 1;  // clamp (dead rows dropped in epilogue)
    v8f acc[4];
    #pragma unroll
    for (int f = 0; f < 4; ++f)
        #pragma unroll
        for (int j = 0; j < 8; ++j) acc[f][j] = 0.0f;

    const int ksteps = K >> 5;
    // prime the pipeline: tile 0 -> buffer 0
    async_ld_b128((unsigned)(unsigned long long)&As[0][r * 40 + c8],
                  A + growc * (i64)K + c8);
    for (int ks = 0; ks < ksteps; ++ks) {
        wait_async0();        // this wave's copy of tile ks has landed in LDS
        __syncthreads();      // everyone's copy landed; prior reads of other buf done
        if (ks + 1 < ksteps)  // overlap copy of tile ks+1 with WMMAs of tile ks
            async_ld_b128((unsigned)(unsigned long long)&As[(ks + 1) & 1][r * 40 + c8],
                          A + growc * (i64)K + (ks + 1) * 32 + c8);
        const __bf16* Ab = &As[ks & 1][0];
        union { v16bf v; bf16x8 h[2]; } fa;
        const int arow  = m0 + (lane & 15);
        const int khalf = (lane >> 4) * 8;
        fa.h[0] = *(const bf16x8*)(Ab + arow * 40 + khalf);
        fa.h[1] = *(const bf16x8*)(Ab + arow * 40 + 16 + khalf);
        #pragma unroll
        for (int f = 0; f < 4; ++f) {
            union { v16bf v; bf16x8 h[2]; } fb;
            const __bf16* bp = Bf + (((i64)(ks * 8 + (n0 >> 4) + f)) * 32 + lane) * 16;
            fb.h[0] = *(const bf16x8*)(bp);
            fb.h[1] = *(const bf16x8*)(bp + 8);
            acc[f] = __builtin_amdgcn_wmma_f32_16x16x32_bf16(
                false, fa.v, false, fb.v, (short)0, acc[f], false, false);
        }
    }
    const int col_l = lane & 15;
    const int moff  = (lane >> 4) * 8;
    #pragma unroll
    for (int f = 0; f < 4; ++f) {
        const int n  = n0 + f * 16 + col_l;
        const float bn = bias[n];
        #pragma unroll
        for (int v = 0; v < 8; ++v) {
            const i64 m = (i64)blockIdx.x * 64 + m0 + moff + v;
            if (m < rows) {
                float val = acc[f][v] + bn;
                i64 idx = m * HID + n;
                if (EPI == 0)      outF[idx] = val;
                else if (EPI == 1) outB[idx] = (__bf16)(val > 0.f ? val : 0.f);
                else if (EPI == 2) outF[idx] = val / (1.f + __expf(-val));
                else               outF[idx] = resid[idx] + val;
            }
        }
    }
}

// --- Message GEMM: A gathered from (h[src] | h[dst] | d), K=384, relu->bf16 -
// f32 sources are converted to bf16 in the LDS fill; the gather loads for the
// next K-chunk are software-pipelined against the current chunk's WMMAs.
__global__ __launch_bounds__(256) void msg_gemm(
    const float* __restrict__ h, const float* __restrict__ d,
    const int* __restrict__ srcI, const int* __restrict__ dstI,
    const __bf16* __restrict__ Bf, const float* __restrict__ bias,
    __bf16* __restrict__ outB, int E) {
    __shared__ alignas(16) __bf16 As[64 * 40];
    const int tid  = threadIdx.x;
    const int wave = tid >> 5, lane = tid & 31;
    const int m0 = (wave >> 1) * 16;
    const int n0 = (wave & 1) * 64;
    const int r  = tid >> 2;
    const int c8 = (tid & 3) * 8;
    const i64 e = (i64)blockIdx.x * 64 + r;
    const bool ok = (e < E);
    const float* hs = ok ? h + (i64)srcI[e] * HID : h;
    const float* hd = ok ? h + (i64)dstI[e] * HID : h;
    const float* de = ok ? d + e * HID : d;
    v8f acc[4];
    #pragma unroll
    for (int f = 0; f < 4; ++f)
        #pragma unroll
        for (int j = 0; j < 8; ++j) acc[f][j] = 0.0f;

    auto loadA = [&](int ks) -> bf16x8 {
        bf16x8 a8;
        if (ok) {
            const int c = ks * 32 + c8;
            const float* sp;
            if (c < 128)      sp = hs + c;
            else if (c < 256) sp = hd + (c - 128);
            else              sp = de + (c - 256);
            f32x4 x0 = *(const f32x4*)sp;
            f32x4 x1 = *(const f32x4*)(sp + 4);
            #pragma unroll
            for (int j = 0; j < 4; ++j) { a8[j] = (__bf16)x0[j]; a8[j + 4] = (__bf16)x1[j]; }
        } else {
            #pragma unroll
            for (int j = 0; j < 8; ++j) a8[j] = (__bf16)0.0f;
        }
        return a8;
    };

    bf16x8 cur = loadA(0);
    for (int ks = 0; ks < 12; ++ks) {           // K = 384
        __syncthreads();
        *(bf16x8*)(&As[r * 40 + c8]) = cur;
        __syncthreads();
        if (ks + 1 < 12) cur = loadA(ks + 1);   // overlap gather with WMMAs

        union { v16bf v; bf16x8 h2[2]; } fa;
        const int arow  = m0 + (lane & 15);
        const int khalf = (lane >> 4) * 8;
        fa.h2[0] = *(const bf16x8*)(&As[arow * 40 + khalf]);
        fa.h2[1] = *(const bf16x8*)(&As[arow * 40 + 16 + khalf]);
        #pragma unroll
        for (int f = 0; f < 4; ++f) {
            union { v16bf v; bf16x8 h2[2]; } fb;
            const __bf16* bp = Bf + (((i64)(ks * 8 + (n0 >> 4) + f)) * 32 + lane) * 16;
            fb.h2[0] = *(const bf16x8*)(bp);
            fb.h2[1] = *(const bf16x8*)(bp + 8);
            acc[f] = __builtin_amdgcn_wmma_f32_16x16x32_bf16(
                false, fa.v, false, fb.v, (short)0, acc[f], false, false);
        }
    }
    const int col_l = lane & 15;
    const int moff  = (lane >> 4) * 8;
    #pragma unroll
    for (int f = 0; f < 4; ++f) {
        const int n  = n0 + f * 16 + col_l;
        const float bn = bias[n];
        #pragma unroll
        for (int v = 0; v < 8; ++v) {
            const i64 m = (i64)blockIdx.x * 64 + m0 + moff + v;
            if (m < E) {
                float val = acc[f][v] + bn;
                outB[m * HID + n] = (__bf16)(val > 0.f ? val : 0.f);
            }
        }
    }
}

// --- misc elementwise / stats kernels ---------------------------------------
__global__ void zero_f32(float* p, i64 n) {
    i64 i = (i64)blockIdx.x * blockDim.x + threadIdx.x;
    i64 st = (i64)gridDim.x * blockDim.x;
    for (; i < n; i += st) p[i] = 0.0f;
}
__global__ void init_h(const float* __restrict__ emb, float* __restrict__ h, i64 n) {
    i64 i = (i64)blockIdx.x * blockDim.x + threadIdx.x;
    if (i < n) h[i] = emb[i & (HID - 1)];
}
__global__ void f32_to_bf16(const float* __restrict__ a, __bf16* __restrict__ o, i64 n) {
    i64 i = (i64)blockIdx.x * blockDim.x + threadIdx.x;
    if (i < n) o[i] = (__bf16)a[i];
}
__global__ void add_to_bf16(const float* __restrict__ a, const float* __restrict__ b,
                            __bf16* __restrict__ o, i64 n) {
    i64 i = (i64)blockIdx.x * blockDim.x + threadIdx.x;
    if (i < n) o[i] = (__bf16)(a[i] + b[i]);
}
__global__ void bn_relu_bf16(const float* __restrict__ z, const float* __restrict__ scale,
                             const float* __restrict__ shift, __bf16* __restrict__ o, i64 n) {
    i64 i = (i64)blockIdx.x * blockDim.x + threadIdx.x;
    if (i < n) {
        int f = (int)(i & (HID - 1));
        float v = z[i] * scale[f] + shift[f];
        o[i] = (__bf16)(v > 0.f ? v : 0.f);
    }
}
// Column stats: partial sums/sumsq over 128-wide rows, atomic into gsum/gsq.
__global__ __launch_bounds__(256) void colstats_partial(
    const float* __restrict__ z, float* __restrict__ gsum, float* __restrict__ gsq, int rows) {
    __shared__ float s1[256], s2[256];
    int f = threadIdx.x & 127, half = threadIdx.x >> 7;
    i64 r = (i64)blockIdx.x * 512 + half;
    float s = 0.f, q = 0.f;
    for (int i = 0; i < 256; ++i, r += 2)
        if (r < rows) { float v = z[r * HID + f]; s += v; q += v * v; }
    s1[threadIdx.x] = s; s2[threadIdx.x] = q;
    __syncthreads();
    if (half == 0) {
        atomicAdd(&gsum[f], s1[f] + s1[f + 128]);
        atomicAdd(&gsq[f],  s2[f] + s2[f + 128]);
    }
}
__global__ void colstats_finalize(const float* __restrict__ gsum, const float* __restrict__ gsq,
                                  const float* __restrict__ g, const float* __restrict__ b,
                                  float* __restrict__ scale, float* __restrict__ shift, int rows) {
    int f = threadIdx.x;  // 128
    float m  = gsum[f] / (float)rows;
    float vr = gsq[f] / (float)rows - m * m;
    float sc = g[f] * rsqrtf(vr + EPS);
    scale[f] = sc;
    shift[f] = b[f] - m * sc;
}
// --- edge feature (fourier) + first 9->128 linear ---------------------------
__global__ __launch_bounds__(128) void edge_in_l1(
    const float* __restrict__ dist, const float* __restrict__ W, const float* __restrict__ b,
    float* __restrict__ z, int E) {
    __shared__ float feats[8][9];
    int t = threadIdx.x;
    i64 e0 = (i64)blockIdx.x * 8;
    if (t < 72) {
        int el = t / 9, j = t % 9;
        i64 e = e0 + el;
        float dv = (e < E) ? dist[e] : 0.f;
        float v;
        if (j < 4)      v = sinf(dv / (float)(1 << j));
        else if (j < 8) v = cosf(dv / (float)(1 << (j - 4)));
        else            v = dv;
        feats[el][j] = v;
    }
    __syncthreads();
    for (int el = 0; el < 8; ++el) {
        i64 e = e0 + el;
        if (e >= E) break;
        float acc = b[t];
        #pragma unroll
        for (int j = 0; j < 9; ++j) acc += feats[el][j] * W[j * HID + t];
        z[e * HID + t] = acc;
    }
}
// --- edge epilogue: gate + d += msg + scatter-add into m_sum ---------------
__global__ __launch_bounds__(256) void edge_epilogue(
    const float* __restrict__ msg, const float* __restrict__ softW, const float* __restrict__ softb,
    const int* __restrict__ dstI, float* __restrict__ d, float* __restrict__ msum, int E) {
    int wave = threadIdx.x >> 5, lane = threadIdx.x & 31;
    i64 e = (i64)blockIdx.x * 8 + wave;
    if (e >= E) return;
    float m[4]; float part = 0.f;
    #pragma unroll
    for (int q = 0; q < 4; ++q) {
        int f = lane + 32 * q;
        m[q] = msg[e * HID + f];
        part += m[q] * softW[f];
    }
    #pragma unroll
    for (int o = 16; o > 0; o >>= 1) part += __shfl_xor(part, o, 32);
    float w = 1.f / (1.f + __expf(-(part + softb[0])));
    i64 db = (i64)dstI[e] * HID;
    #pragma unroll
    for (int q = 0; q < 4; ++q) {
        int f = lane + 32 * q;
        d[e * HID + f] += m[q];
        atomicAdd(&msum[db + f], m[q] * w);
    }
}
// --- readout over sorted segment ids ---------------------------------------
__global__ __launch_bounds__(128) void readout_kernel(
    const float* __restrict__ hout, const int* __restrict__ gid,
    float* __restrict__ ro, int N, int G) {
    int g = blockIdx.x, f = threadIdx.x;
    int lo, hi;
    { int a = 0, b = N; while (a < b) { int mm = (a + b) >> 1; if (gid[mm] < g) a = mm + 1; else b = mm; } lo = a; }
    { int a = lo, b = N; while (a < b) { int mm = (a + b) >> 1; if (gid[mm] < g + 1) a = mm + 1; else b = mm; } hi = a; }
    float s = 0.f, mx = -3.4e38f;
    for (int i = lo; i < hi; ++i) {
        float v = hout[(i64)i * HID + f];
        s += v; mx = v > mx ? v : mx;
    }
    int cnt = hi - lo;
    ro[(i64)g * 384 + f]       = s;
    ro[(i64)g * 384 + 128 + f] = cnt ? (s / (float)cnt) : 0.f;
    ro[(i64)g * 384 + 256 + f] = cnt ? mx : 0.f;
}
// --- tiny output MLP --------------------------------------------------------
__global__ __launch_bounds__(128) void out_l1(const float* __restrict__ ro, const float* __restrict__ W,
                                              const float* __restrict__ b, float* __restrict__ zo, int G) {
    __shared__ float row[384];
    int g = blockIdx.x, t = threadIdx.x;
    for (int j = t; j < 384; j += 128) row[j] = ro[(i64)g * 384 + j];
    __syncthreads();
    float acc = b[t];
    for (int j = 0; j < 384; ++j) acc += row[j] * W[(i64)j * HID + t];
    zo[(i64)g * HID + t] = acc;
}
__global__ __launch_bounds__(128) void out_l2(const float* __restrict__ zo, const float* __restrict__ scale,
                                              const float* __restrict__ shift, const float* __restrict__ W2,
                                              const float* __restrict__ b2, float* __restrict__ out, int G) {
    __shared__ float red[128];
    int g = blockIdx.x, t = threadIdx.x;
    float v = zo[(i64)g * HID + t] * scale[t] + shift[t];
    v = v > 0.f ? v : 0.f;
    red[t] = v * W2[t];
    __syncthreads();
    for (int s = 64; s > 0; s >>= 1) { if (t < s) red[t] += red[t + s]; __syncthreads(); }
    if (t == 0) out[g] = red[0] + b2[0];
}

// ---------------------------------------------------------------------------
extern "C" void kernel_launch(void* const* d_in, const int* in_sizes, int n_in,
                              void* d_out, int out_size, void* d_ws, size_t ws_size,
                              hipStream_t stream) {
    // JAX pytree leaf order (dict keys sorted: ASCII => 'Ws' < 'bn' < 'bs';
    // layers: 'message' < 'soft_W' < 'soft_b' < 'update').
    const float* EI_W0 = (const float*)d_in[0];   // (9,128)
    const float* EI_W1 = (const float*)d_in[1];   // (128,128)
    const float* EI_g  = (const float*)d_in[2];
    const float* EI_bb = (const float*)d_in[3];
    const float* EI_b0 = (const float*)d_in[4];
    const float* EI_b1 = (const float*)d_in[5];
    const float* node_emb = (const float*)d_in[54];
    const float* NO_W0 = (const float*)d_in[55];
    const float* NO_W1 = (const float*)d_in[56];
    const float* NO_g  = (const float*)d_in[57];
    const float* NO_bb = (const float*)d_in[58];
    const float* NO_b0 = (const float*)d_in[59];
    const float* NO_b1 = (const float*)d_in[60];
    const float* OU_W0 = (const float*)d_in[61];  // (384,128)
    const float* OU_W1 = (const float*)d_in[62];  // (128,1)
    const float* OU_g  = (const float*)d_in[63];
    const float* OU_bb = (const float*)d_in[64];
    const float* OU_b0 = (const float*)d_in[65];
    const float* OU_b1 = (const float*)d_in[66];
    const float* dist  = (const float*)d_in[67];
    const int*   srcI  = (const int*)d_in[68];
    const int*   dstI  = (const int*)d_in[69];
    const int*   gid   = (const int*)d_in[70];

    const int E = in_sizes[67];
    const int N = in_sizes[70];
    const int G = out_size;
    const i64 En = (i64)E * HID, Nn = (i64)N * HID;

    // workspace bump allocator
    char* wp = (char*)d_ws;
    auto alloc = [&](size_t bytes) -> void* {
        void* p = (void*)wp;
        wp += (bytes + 255) & ~(size_t)255;
        return p;
    };
    float*  dF    = (float*)alloc(En * 4);       // edge features d  [E,128]
    float*  ebufF = (float*)alloc(En * 4);       // z_e / msg        [E,128]
    __bf16* ebufB = (__bf16*)alloc(En * 2);      // bf16 activations [E,128]
    float*  h     = (float*)alloc(Nn * 4);
    float*  msum  = (float*)alloc(Nn * 4);       // reused as hout at the end
    float*  zn    = (float*)alloc(Nn * 4);
    __bf16* nb    = (__bf16*)alloc(Nn * 2);
    float*  ro    = (float*)alloc((i64)G * 384 * 4);
    float*  zo    = (float*)alloc((i64)G * HID * 4);
    float*  gsum  = (float*)alloc(HID * 4);
    float*  gsq   = (float*)alloc(HID * 4);
    float*  scale = (float*)alloc(HID * 4);
    float*  shift = (float*)alloc(HID * 4);
    __bf16* wEI1  = (__bf16*)alloc((i64)128 * HID * 2);
    __bf16 *wM0[4], *wM1[4], *wU0[4], *wU1[4];
    for (int l = 0; l < 4; ++l) {
        wM0[l] = (__bf16*)alloc((i64)384 * HID * 2);
        wM1[l] = (__bf16*)alloc((i64)128 * HID * 2);
        wU0[l] = (__bf16*)alloc((i64)128 * HID * 2);
        wU1[l] = (__bf16*)alloc((i64)128 * HID * 2);
    }
    __bf16* wNO0 = (__bf16*)alloc((i64)128 * HID * 2);
    __bf16* wNO1 = (__bf16*)alloc((i64)128 * HID * 2);
    (void)ws_size; (void)n_in;

    const int eb64 = (E + 63) / 64, nb64 = (N + 63) / 64;
    const int eElt = (int)((En + 255) / 256), nElt = (int)((Nn + 255) / 256);

    // 1. weight conversion to fragment layout
    wconv_frag<<<(128 * 8 + 255) / 256, 256, 0, stream>>>(EI_W1, wEI1, 128);
    for (int l = 0; l < 4; ++l) {
        int base = 6 + 12 * l;
        wconv_frag<<<(384 * 8 + 255) / 256, 256, 0, stream>>>((const float*)d_in[base + 0], wM0[l], 384);
        wconv_frag<<<(128 * 8 + 255) / 256, 256, 0, stream>>>((const float*)d_in[base + 1], wM1[l], 128);
        wconv_frag<<<(128 * 8 + 255) / 256, 256, 0, stream>>>((const float*)d_in[base + 6], wU0[l], 128);
        wconv_frag<<<(128 * 8 + 255) / 256, 256, 0, stream>>>((const float*)d_in[base + 7], wU1[l], 128);
    }
    wconv_frag<<<(128 * 8 + 255) / 256, 256, 0, stream>>>(NO_W0, wNO0, 128);
    wconv_frag<<<(128 * 8 + 255) / 256, 256, 0, stream>>>(NO_W1, wNO1, 128);

    // 2. node init + edge input network
    init_h<<<nElt, 256, 0, stream>>>(node_emb, h, Nn);
    edge_in_l1<<<(E + 7) / 8, 128, 0, stream>>>(dist, EI_W0, EI_b0, ebufF, E);
    zero_f32<<<1, 256, 0, stream>>>(gsum, HID);
    zero_f32<<<1, 256, 0, stream>>>(gsq, HID);
    colstats_partial<<<(E + 511) / 512, 256, 0, stream>>>(ebufF, gsum, gsq, E);
    colstats_finalize<<<1, 128, 0, stream>>>(gsum, gsq, EI_g, EI_bb, scale, shift, E);
    bn_relu_bf16<<<eElt, 256, 0, stream>>>(ebufF, scale, shift, ebufB, En);
    gemm_bf16<2><<<eb64, 256, 0, stream>>>(ebufB, wEI1, EI_b1, nullptr, dF, nullptr, E, 128);

    // 3. message-passing layers
    for (int l = 0; l < 4; ++l) {
        int base = 6 + 12 * l;
        const float* bm0 = (const float*)d_in[base + 2];
        const float* bm1 = (const float*)d_in[base + 3];
        const float* sW  = (const float*)d_in[base + 4];
        const float* sb  = (const float*)d_in[base + 5];
        const float* ug  = (const float*)d_in[base + 8];
        const float* ub  = (const float*)d_in[base + 9];
        const float* bu0 = (const float*)d_in[base + 10];
        const float* bu1 = (const float*)d_in[base + 11];

        msg_gemm<<<eb64, 256, 0, stream>>>(h, dF, srcI, dstI, wM0[l], bm0, ebufB, E);
        gemm_bf16<2><<<eb64, 256, 0, stream>>>(ebufB, wM1[l], bm1, nullptr, ebufF, nullptr, E, 128);
        zero_f32<<<4096, 256, 0, stream>>>(msum, Nn);
        edge_epilogue<<<(E + 7) / 8, 256, 0, stream>>>(ebufF, sW, sb, dstI, dF, msum, E);
        add_to_bf16<<<nElt, 256, 0, stream>>>(msum, h, nb, Nn);
        gemm_bf16<0><<<nb64, 256, 0, stream>>>(nb, wU0[l], bu0, nullptr, zn, nullptr, N, 128);
        zero_f32<<<1, 256, 0, stream>>>(gsum, HID);
        zero_f32<<<1, 256, 0, stream>>>(gsq, HID);
        colstats_partial<<<(N + 511) / 512, 256, 0, stream>>>(zn, gsum, gsq, N);
        colstats_finalize<<<1, 128, 0, stream>>>(gsum, gsq, ug, ub, scale, shift, N);
        bn_relu_bf16<<<nElt, 256, 0, stream>>>(zn, scale, shift, nb, Nn);
        gemm_bf16<3><<<nb64, 256, 0, stream>>>(nb, wU1[l], bu1, h, h, nullptr, N, 128);
    }

    // 4. node output network (hout stored in msum buffer)
    f32_to_bf16<<<nElt, 256, 0, stream>>>(h, nb, Nn);
    gemm_bf16<0><<<nb64, 256, 0, stream>>>(nb, wNO0, NO_b0, nullptr, zn, nullptr, N, 128);
    zero_f32<<<1, 256, 0, stream>>>(gsum, HID);
    zero_f32<<<1, 256, 0, stream>>>(gsq, HID);
    colstats_partial<<<(N + 511) / 512, 256, 0, stream>>>(zn, gsum, gsq, N);
    colstats_finalize<<<1, 128, 0, stream>>>(gsum, gsq, NO_g, NO_bb, scale, shift, N);
    bn_relu_bf16<<<nElt, 256, 0, stream>>>(zn, scale, shift, nb, Nn);
    gemm_bf16<0><<<nb64, 256, 0, stream>>>(nb, wNO1, NO_b1, nullptr, msum, nullptr, N, 128);

    // 5. readout + output MLP
    readout_kernel<<<G, 128, 0, stream>>>(msum, gid, ro, N, G);
    out_l1<<<G, 128, 0, stream>>>(ro, OU_W0, OU_b0, zo, G);
    zero_f32<<<1, 256, 0, stream>>>(gsum, HID);
    zero_f32<<<1, 256, 0, stream>>>(gsq, HID);
    colstats_partial<<<(G + 511) / 512, 256, 0, stream>>>(zo, gsum, gsq, G);
    colstats_finalize<<<1, 128, 0, stream>>>(gsum, gsq, OU_g, OU_bb, scale, shift, G);
    out_l2<<<G, 128, 0, stream>>>(zo, scale, shift, OU_W1, OU_b1, (float*)d_out, G);
}
